// FrameTransformer_74534862455221
// MI455X (gfx1250) — matrix operand order repaired and checked by reading, not compiled, not run on hardware
//
#include <hip/hip_runtime.h>

#define C_CH   8
#define F_DIM  1024
#define W_SEQ  512
#define H_HEADS 16
#define D_HEAD 64
#define FH_DIM 4096   // F * EXP

typedef __attribute__((ext_vector_type(16))) __bf16 v16bf;
typedef __attribute__((ext_vector_type(8)))  float  v8f;
typedef __attribute__((ext_vector_type(4)))  int    v4i;

#if defined(__has_builtin)
#if __has_builtin(__builtin_amdgcn_global_load_async_to_lds_b128)
#define HAVE_ASYNC_LDS 1
#endif
#if __has_builtin(__builtin_amdgcn_s_wait_asynccnt)
#define HAVE_WAIT_ASYNC 1
#endif
#endif

union FragV {
  uint4 q[2];
  v16bf v;
};

union H8 {
  unsigned short h[8];
  uint4 u;
};

__device__ __forceinline__ unsigned short f2bf(float f) {
  unsigned u = __float_as_uint(f);
  u = (u + 0x7fffu + ((u >> 16) & 1u)) >> 16;   // round-to-nearest-even
  return (unsigned short)u;
}

__device__ __forceinline__ v8f wmma_bf16(const FragV& a, const FragV& b, v8f c) {
  // D = A(16x32 bf16) * B(32x16 bf16) + C(16x16 f32)
  return __builtin_amdgcn_wmma_f32_16x16x32_bf16(
      /*neg_a=*/false, a.v, /*neg_b=*/false, b.v,
      /*c_mod=*/(short)0, c, /*reuse_a=*/false, /*reuse_b=*/false);
}

// 16-byte global -> LDS copy: async (ASYNCcnt-tracked) when available.
__device__ __forceinline__ void cp16_to_lds(unsigned short* lds_dst,
                                            const unsigned short* gsrc) {
#ifdef HAVE_ASYNC_LDS
  __builtin_amdgcn_global_load_async_to_lds_b128(
      (__attribute__((address_space(1))) v4i*)(gsrc),
      (__attribute__((address_space(3))) v4i*)(lds_dst),
      /*offset=*/0, /*cpol=*/0);
#else
  *(uint4*)lds_dst = *(const uint4*)gsrc;
#endif
}

__device__ __forceinline__ void wait_async_copies() {
#ifdef HAVE_ASYNC_LDS
#ifdef HAVE_WAIT_ASYNC
  __builtin_amdgcn_s_wait_asynccnt(0);
#else
  asm volatile("s_wait_asynccnt 0x0" ::: "memory");
#endif
#endif
}

// ---------------------------------------------------------------------------
// fp32 -> bf16 elementwise convert (weights). n must be a multiple of 4.
// ---------------------------------------------------------------------------
__global__ __launch_bounds__(256) void cvt_bf16_kernel(
    const float* __restrict__ src, unsigned short* __restrict__ dst, long long n4) {
  long long i = (long long)blockIdx.x * 256 + threadIdx.x;
  if (i >= n4) return;
  float4 v = ((const float4*)src)[i];
  unsigned lo = (unsigned)f2bf(v.x) | ((unsigned)f2bf(v.y) << 16);
  unsigned hi = (unsigned)f2bf(v.z) | ((unsigned)f2bf(v.w) << 16);
  ((uint2*)dst)[i] = make_uint2(lo, hi);
}

// ---------------------------------------------------------------------------
// frame_norm over the F axis of X[C][F][W]; writes Z transposed: Zt[C][W][F] bf16
// ---------------------------------------------------------------------------
__global__ __launch_bounds__(256) void frame_norm_kernel(
    const float* __restrict__ X, const float* __restrict__ g,
    const float* __restrict__ b, unsigned short* __restrict__ Zt) {
  int c  = blockIdx.y;
  int w0 = blockIdx.x * 32;
  int tx = threadIdx.x & 31;       // w lane
  int ty = threadIdx.x >> 5;       // 0..7
  const float* Xc = X + (long long)c * F_DIM * W_SEQ;

  float s1 = 0.f, s2 = 0.f;
  for (int f = ty; f < F_DIM; f += 8) {
    float v = Xc[(long long)f * W_SEQ + w0 + tx];
    s1 += v; s2 += v * v;
  }
  __shared__ float r1[8][32], r2[8][32];
  r1[ty][tx] = s1; r2[ty][tx] = s2;
  __syncthreads();
  float t1 = 0.f, t2 = 0.f;
#pragma unroll
  for (int i = 0; i < 8; ++i) { t1 += r1[i][tx]; t2 += r2[i][tx]; }
  float mu  = t1 * (1.0f / F_DIM);
  float var = t2 * (1.0f / F_DIM) - mu * mu;
  float rs  = rsqrtf(var + 1e-5f);

  unsigned short* zr = Zt + ((long long)c * W_SEQ + w0 + tx) * F_DIM;
  for (int f = ty * 128; f < ty * 128 + 128; f += 8) {
    H8 h8;
#pragma unroll
    for (int i = 0; i < 8; ++i) {
      float v = Xc[(long long)(f + i) * W_SEQ + w0 + tx];
      v = (v - mu) * rs * g[f + i] + b[f + i];
      h8.h[i] = f2bf(v);
    }
    *(uint4*)&zr[f] = h8.u;
  }
}

// ---------------------------------------------------------------------------
// Generic per-channel bf16 GEMM:  D[c] = A[c](MxK) * Bt[c](NxK)^T  (+Cres)
// MODE 0: f32 out [c][M][N] (+ optional residual)
// MODE 1: exact-GELU epilogue, bf16 transposed out [c][N][M]
// Block 256 thr = 8 waves; block tile 128x64; wave tile 32x32 (2x2 WMMA).
// Double-buffered LDS, filled with GLOBAL_LOAD_ASYNC_TO_LDS_B128.
// ---------------------------------------------------------------------------
template <int MODE>
__global__ __launch_bounds__(256) void gemm_bf16_kernel(
    const unsigned short* __restrict__ A,
    const unsigned short* __restrict__ Bt,
    const float* __restrict__ Cres,
    float* __restrict__ Df,
    unsigned short* __restrict__ Dt,
    int M, int N, int K) {
  int c = blockIdx.z;
  A  += (long long)c * M * K;
  Bt += (long long)c * N * K;
  long long mn = (long long)M * N;
  if (Cres) Cres += (long long)c * mn;
  if (MODE == 0) Df += (long long)c * mn;
  else           Dt += (long long)c * mn;

  __shared__ unsigned short As[2][128][32];   // 2 x 8 KB
  __shared__ unsigned short Bs[2][64][32];    // 2 x 4 KB

  int m0 = blockIdx.y * 128;
  int n0 = blockIdx.x * 64;
  int tid  = threadIdx.x;
  int wave = tid >> 5;
  int lane = tid & 31;
  int half = lane >> 4;        // 0: lanes 0-15, 1: lanes 16-31
  int l16  = lane & 15;
  int wm = (wave & 3) * 32;    // wave m offset within block tile
  int wn = (wave >> 2) * 32;   // wave n offset

  int rr = tid >> 2;           // 0..63: chunk row for A (two rows) and Bt
  int kc = (tid & 3) * 8;      // chunk k offset (elements)

  auto stage = [&](int k0, int buf) {
#pragma unroll
    for (int i = 0; i < 2; ++i) {
      int r = rr + i * 64;     // A rows 0..127
      cp16_to_lds(&As[buf][r][kc], A + (long long)(m0 + r) * K + k0 + kc);
    }
    cp16_to_lds(&Bs[buf][rr][kc], Bt + (long long)(n0 + rr) * K + k0 + kc);
  };

  v8f acc[2][2] = {};
  stage(0, 0);
  int buf = 0;

  for (int k0 = 0; k0 < K; k0 += 32) {
    wait_async_copies();       // own async copies into `buf` landed
    __syncthreads();           // everyone's copies visible
    if (k0 + 32 < K) stage(k0 + 32, buf ^ 1);
    if (k0 + 64 < K)           // global_prefetch_b8 two tiles ahead
      __builtin_prefetch(A + (long long)(m0 + rr) * K + k0 + 64, 0, 1);

    FragV a[2], b[2];
    int kbase = half * 8;      // A layout: chunks at kbase, kbase+16
#pragma unroll
    for (int i = 0; i < 2; ++i) {
      int r = wm + i * 16 + l16;
      a[i].q[0] = *(const uint4*)&As[buf][r][kbase];
      a[i].q[1] = *(const uint4*)&As[buf][r][kbase + 16];
    }
#pragma unroll
    for (int j = 0; j < 2; ++j) {
      int r  = wn + j * 16 + l16;
      int kb = half * 8;       // B layout: chunks at kb, kb+16 (same as A)
      b[j].q[0] = *(const uint4*)&Bs[buf][r][kb];
      b[j].q[1] = *(const uint4*)&Bs[buf][r][kb + 16];
    }
#pragma unroll
    for (int i = 0; i < 2; ++i)
#pragma unroll
      for (int j = 0; j < 2; ++j)
        acc[i][j] = wmma_bf16(a[i], b[j], acc[i][j]);
    buf ^= 1;
  }

  // epilogue
#pragma unroll
  for (int i = 0; i < 2; ++i)
#pragma unroll
    for (int j = 0; j < 2; ++j) {
      int mbase = m0 + wm + i * 16 + half * 8;  // 8 consecutive rows per lane
      int nn    = n0 + wn + j * 16 + l16;
      if (MODE == 0) {
#pragma unroll
        for (int r = 0; r < 8; ++r) {
          float v = acc[i][j][r];
          long long idx = (long long)(mbase + r) * N + nn;
          if (Cres) v += Cres[idx];
          Df[idx] = v;
        }
      } else {
        H8 h8;
#pragma unroll
        for (int r = 0; r < 8; ++r) {
          float v = acc[i][j][r];
          v = 0.5f * v * (1.0f + erff(v * 0.70710678f));   // exact GELU
          h8.h[r] = f2bf(v);
        }
        *(uint4*)&Dt[(long long)nn * M + mbase] = h8.u;    // transposed bf16 store
      }
    }
}

// ---------------------------------------------------------------------------
// RoPE + layout split:  P[C][F][W] -> Q[C][H][W][64] (rotated, bf16; serves q AND k
// since reference applies identical RoPE to both) and Vt[C][H][64][W] (pre-rope).
// ---------------------------------------------------------------------------
__global__ __launch_bounds__(256) void rope_kernel(
    const float* __restrict__ P, const float* __restrict__ freqs,
    unsigned short* __restrict__ Qb, unsigned short* __restrict__ Vt) {
  int gid = blockIdx.x * 256 + threadIdx.x;   // C*H*W threads
  int w  = gid & (W_SEQ - 1);
  int ch = gid >> 9;                          // c*H + h
  const float* pp = P + (long long)ch * 64 * W_SEQ + w;

  float t[64];
#pragma unroll
  for (int d = 0; d < 64; ++d) t[d] = pp[(long long)d * W_SEQ];

  unsigned short qb[64];
  float wp = (float)w;
#pragma unroll
  for (int i = 0; i < 16; ++i) {              // ROPE_DIM=32 -> 16 rotation pairs
    float th = wp * freqs[i];
    float sn, cs;
    __sincosf(th, &sn, &cs);
    qb[2 * i]     = f2bf(t[2 * i] * cs - t[2 * i + 1] * sn);
    qb[2 * i + 1] = f2bf(t[2 * i + 1] * cs + t[2 * i] * sn);
  }
#pragma unroll
  for (int d = 32; d < 64; ++d) qb[d] = f2bf(t[d]);   // pass-through dims

  uint4* qdst = (uint4*)(Qb + ((long long)ch * W_SEQ + w) * 64);
#pragma unroll
  for (int i = 0; i < 8; ++i) qdst[i] = ((uint4*)qb)[i];

  unsigned short* vt = Vt + (long long)ch * 64 * W_SEQ + w;
#pragma unroll
  for (int d = 0; d < 64; ++d) vt[(long long)d * W_SEQ] = f2bf(t[d]);
}

// ---------------------------------------------------------------------------
// Flash attention, one wave (32 thr) per (c, h, 16-row q block).
// q == k (identical RoPE).  Writes O transposed: Ot[C][W][F], f = h*64+d.
// ---------------------------------------------------------------------------
__global__ __launch_bounds__(32) void attn_kernel(
    const unsigned short* __restrict__ Q,   // [C][H][W][64]
    const unsigned short* __restrict__ Vt,  // [C][H][64][W]
    unsigned short* __restrict__ Ot) {      // [C][W][F]
  int qb = blockIdx.x;                      // 0..W/16-1
  int h  = blockIdx.y;
  int c  = blockIdx.z;
  __shared__ unsigned short Pl[16][32];     // P tile staging (C-layout -> A-layout)

  int lane = threadIdx.x;
  int half = lane >> 4;
  int l16  = lane & 15;
  int kb   = half * 8;

  const unsigned short* Qh = Q  + (long long)(c * H_HEADS + h) * W_SEQ * 64;
  const unsigned short* Vh = Vt + (long long)(c * H_HEADS + h) * 64 * W_SEQ;
  int q0 = qb * 16;

  // Q fragments (A layout), d halves 0..31 / 32..63
  FragV qa[2];
  {
    const unsigned short* qr = Qh + (long long)(q0 + l16) * 64;
    qa[0].q[0] = *(const uint4*)(qr + kb);
    qa[0].q[1] = *(const uint4*)(qr + kb + 16);
    qa[1].q[0] = *(const uint4*)(qr + 32 + kb);
    qa[1].q[1] = *(const uint4*)(qr + 32 + kb + 16);
  }

  float mrow[8], lrow[8];
#pragma unroll
  for (int r = 0; r < 8; ++r) { mrow[r] = -1e30f; lrow[r] = 0.f; }
  v8f o[4] = {};
  const float scale = 1.0f / 32.0f;         // 1/sqrt(F), F=1024

  for (int kk = 0; kk < W_SEQ; kk += 32) {
    // ---- S = Q K^T for two 16-wide k tiles
    v8f s[2];
#pragma unroll
    for (int t = 0; t < 2; ++t) {
      const unsigned short* kr = Qh + (long long)(kk + t * 16 + l16) * 64;
      FragV b0, b1;
      b0.q[0] = *(const uint4*)(kr + half * 8);
      b0.q[1] = *(const uint4*)(kr + half * 8 + 16);
      b1.q[0] = *(const uint4*)(kr + 32 + half * 8);
      b1.q[1] = *(const uint4*)(kr + 32 + half * 8 + 16);
      v8f z = {};
      z = wmma_bf16(qa[0], b0, z);
      z = wmma_bf16(qa[1], b1, z);
#pragma unroll
      for (int r = 0; r < 8; ++r) s[t][r] = z[r] * scale;
    }

    // ---- online softmax (rows split across lane halves; reduce within 16 lanes)
#pragma unroll
    for (int r = 0; r < 8; ++r) {
      float bm = fmaxf(s[0][r], s[1][r]);
#pragma unroll
      for (int m = 8; m >= 1; m >>= 1) bm = fmaxf(bm, __shfl_xor(bm, m, 32));
      float nm   = fmaxf(mrow[r], bm);
      float corr = __expf(mrow[r] - nm);
      mrow[r] = nm;
      float p0 = __expf(s[0][r] - nm);
      float p1 = __expf(s[1][r] - nm);
      float ps = p0 + p1;
#pragma unroll
      for (int m = 8; m >= 1; m >>= 1) ps += __shfl_xor(ps, m, 32);
      lrow[r] = lrow[r] * corr + ps;
#pragma unroll
      for (int dt = 0; dt < 4; ++dt) o[dt][r] *= corr;
      int rr = r + half * 8;                 // C-layout row for this lane half
      Pl[rr][l16]      = f2bf(p0);
      Pl[rr][16 + l16] = f2bf(p1);
    }
    __syncthreads();                         // order LDS stores before A-frag reads

    // ---- O += P V   (P from LDS in A-layout, V transposed: contiguous in k)
    FragV pa;
    pa.q[0] = *(const uint4*)&Pl[l16][kb];
    pa.q[1] = *(const uint4*)&Pl[l16][kb + 16];
#pragma unroll
    for (int dt = 0; dt < 4; ++dt) {
      const unsigned short* vr = Vh + (long long)(dt * 16 + l16) * W_SEQ + kk + half * 8;
      FragV vb;
      vb.q[0] = *(const uint4*)vr;
      vb.q[1] = *(const uint4*)(vr + 16);
      o[dt] = wmma_bf16(pa, vb, o[dt]);
    }
    __syncthreads();
  }

  // ---- normalize, store transposed bf16
#pragma unroll
  for (int r = 0; r < 8; ++r) {
    float inv = 1.0f / lrow[r];
    int w = q0 + r + half * 8;
    unsigned short* orow = Ot + ((long long)c * W_SEQ + w) * F_DIM + h * 64;
#pragma unroll
    for (int dt = 0; dt < 4; ++dt) orow[dt * 16 + l16] = f2bf(o[dt][r] * inv);
  }
}

// ---------------------------------------------------------------------------
extern "C" void kernel_launch(void* const* d_in, const int* in_sizes, int n_in,
                              void* d_out, int out_size, void* d_ws, size_t ws_size,
                              hipStream_t stream) {
  const float* x     = (const float*)d_in[0];
  const float* n1g   = (const float*)d_in[1];
  const float* n1b   = (const float*)d_in[2];
  const float* wq    = (const float*)d_in[3];
  const float* wo    = (const float*)d_in[4];
  const float* freqs = (const float*)d_in[5];
  const float* n2g   = (const float*)d_in[6];
  const float* n2b   = (const float*)d_in[7];
  const float* w1    = (const float*)d_in[8];
  const float* w2    = (const float*)d_in[9];

  char* ws = (char*)d_ws;
  size_t off = 0;
  auto alloc = [&](size_t bytes) { size_t r = off; off += (bytes + 255) & ~(size_t)255; return r; };

  const long long NWQ = (long long)C_CH * F_DIM * F_DIM;     // 8M
  const long long NW1 = (long long)C_CH * FH_DIM * F_DIM;    // 33.5M
  const long long NACT = (long long)C_CH * F_DIM * W_SEQ;    // 4M

  unsigned short* wq_bf = (unsigned short*)(ws + alloc(NWQ * 2));
  unsigned short* wo_bf = (unsigned short*)(ws + alloc(NWQ * 2));
  unsigned short* w1_bf = (unsigned short*)(ws + alloc(NW1 * 2));
  unsigned short* w2_bf = (unsigned short*)(ws + alloc(NW1 * 2));
  unsigned short* z_t   = (unsigned short*)(ws + alloc(NACT * 2));  // reused for norm2
  float*          p     = (float*)(ws + alloc(NACT * 4));
  unsigned short* q_bf  = (unsigned short*)(ws + alloc(NACT * 2));
  unsigned short* v_t   = (unsigned short*)(ws + alloc(NACT * 2));
  unsigned short* o_t   = (unsigned short*)(ws + alloc(NACT * 2));
  float*          x2    = (float*)(ws + alloc(NACT * 4));
  unsigned short* h_t   = (unsigned short*)(ws + alloc((long long)C_CH * W_SEQ * FH_DIM * 2));

  // weight fp32 -> bf16
  cvt_bf16_kernel<<<(unsigned)(NWQ / 4 / 256), 256, 0, stream>>>(wq, wq_bf, NWQ / 4);
  cvt_bf16_kernel<<<(unsigned)(NWQ / 4 / 256), 256, 0, stream>>>(wo, wo_bf, NWQ / 4);
  cvt_bf16_kernel<<<(unsigned)(NW1 / 4 / 256), 256, 0, stream>>>(w1, w1_bf, NW1 / 4);
  cvt_bf16_kernel<<<(unsigned)(NW1 / 4 / 256), 256, 0, stream>>>(w2, w2_bf, NW1 / 4);

  // z = frame_norm(x)  (transposed bf16)
  frame_norm_kernel<<<dim3(W_SEQ / 32, C_CH), 256, 0, stream>>>(x, n1g, n1b, z_t);

  // p = Wq . z
  gemm_bf16_kernel<0><<<dim3(W_SEQ / 64, F_DIM / 128, C_CH), 256, 0, stream>>>(
      wq_bf, z_t, nullptr, p, nullptr, F_DIM, W_SEQ, F_DIM);

  // RoPE + q/v layout split
  rope_kernel<<<(C_CH * H_HEADS * W_SEQ) / 256, 256, 0, stream>>>(p, freqs, q_bf, v_t);

  // flash attention -> o_t (transposed bf16)
  attn_kernel<<<dim3(W_SEQ / 16, H_HEADS, C_CH), 32, 0, stream>>>(q_bf, v_t, o_t);

  // x2 = x + Wo . o
  gemm_bf16_kernel<0><<<dim3(W_SEQ / 64, F_DIM / 128, C_CH), 256, 0, stream>>>(
      wo_bf, o_t, x, x2, nullptr, F_DIM, W_SEQ, F_DIM);

  // z2 = frame_norm(x2)  (reuse z_t)
  frame_norm_kernel<<<dim3(W_SEQ / 32, C_CH), 256, 0, stream>>>(x2, n2g, n2b, z_t);

  // h = gelu(w1 . z2)  (transposed bf16 out)
  gemm_bf16_kernel<1><<<dim3(W_SEQ / 64, FH_DIM / 128, C_CH), 256, 0, stream>>>(
      w1_bf, z_t, nullptr, nullptr, h_t, FH_DIM, W_SEQ, F_DIM);

  // out = x2 + w2 . h
  gemm_bf16_kernel<0><<<dim3(W_SEQ / 64, F_DIM / 128, C_CH), 256, 0, stream>>>(
      w2_bf, h_t, x2, (float*)d_out, nullptr, F_DIM, W_SEQ, FH_DIM);
}
